// DHCF_71897752535221
// MI455X (gfx1250) — matrix operand order repaired and checked by reading, not compiled, not run on hardware
//
#include <hip/hip_runtime.h>
#include <math.h>

// ---------------------------------------------------------------------------
// DHCF on MI455X (gfx1250): everything factors into 24 skinny GEMMs
//   C[R x 64] = A[R x K] * B[K x 64]
// with A = H (binary, exact in f16) or H^T, executed with
// V_WMMA_F32_16X16X32_F16 (f32 accumulate). A is pre-swizzled once into the
// WMMA A-fragment layout; B (tiny, <=8192x64) is re-swizzled per GEMM with
// the diagonal row-scales fused in.
// ---------------------------------------------------------------------------

typedef __attribute__((ext_vector_type(16))) _Float16 v16h;
typedef __attribute__((ext_vector_type(8)))  float    v8f;

#define NU 8192
#define NI 4096
#define DE 64
#define EPS 1e-7f

// ---------------------------------------------------------------------------
// Build A in WMMA 16-bit A-fragment order (16x32 tiles).
// Per ISA 7.12.2: lane l (h=l>>4, r=rt*16+(l&15)) holds 16 f16:
//   elem j<8 : k = kb*32 + h*8 + j
//   elem j>=8: k = kb*32 + 16 + h*8 + (j-8)
// Storage: Asw[(((rt*KS)+kb)*32 + l)*16 + j]  -> 32B contiguous per lane/tile.
// ---------------------------------------------------------------------------
__global__ void make_a_frag(const float* __restrict__ S, int R, int K, int ld,
                            int transposed, _Float16* __restrict__ Asw) {
    size_t tid = (size_t)blockIdx.x * 256 + threadIdx.x;
    size_t total = (size_t)R * K;
    if (tid >= total) return;
    int j  = (int)(tid & 15);
    int l  = (int)((tid >> 4) & 31);
    size_t t2 = tid >> 9;                 // rt*KS + kb
    int KS = K >> 5;
    int kb = (int)(t2 % KS);
    int rt = (int)(t2 / KS);
    int h  = l >> 4;
    int r  = rt * 16 + (l & 15);
    int k  = kb * 32 + ((j < 8) ? (h * 8 + j) : (16 + h * 8 + (j - 8)));
    float v = transposed ? S[(size_t)k * ld + r] : S[(size_t)r * ld + k];
    Asw[tid] = (_Float16)v;
}

// ---------------------------------------------------------------------------
// Build B fragment (32x16 tiles): lane l (h=l>>4, n=ct*16+(l&15)) holds
//   elem j<8 : k = kb*32 + h*8 + j ; elem j>=8: k = kb*32 + 16 + h*8 + (j-8)
// Fused: val = (s ? s[k] : 1) * X[k,n] + (Y ? Y[k,n] : 0)
// Storage: Bsw[(((kb*4)+ct)*32 + l)*16 + j]
// ---------------------------------------------------------------------------
__global__ void make_b_frag(const float* __restrict__ X, const float* __restrict__ Y,
                            const float* __restrict__ s, int K,
                            _Float16* __restrict__ Bsw) {
    size_t tid = (size_t)blockIdx.x * 256 + threadIdx.x;
    size_t total = (size_t)K * 64;
    if (tid >= total) return;
    int j  = (int)(tid & 15);
    int l  = (int)((tid >> 4) & 31);
    size_t t2 = tid >> 9;                 // kb*4 + ct
    int ct = (int)(t2 & 3);
    int kb = (int)(t2 >> 2);
    int h  = l >> 4;
    int n  = ct * 16 + (l & 15);
    int k  = kb * 32 + ((j < 8) ? (h * 8 + j) : (16 + h * 8 + (j - 8)));
    float v = X[(size_t)k * 64 + n];
    if (s) v *= s[k];
    if (Y) v += Y[(size_t)k * 64 + n];
    Bsw[tid] = (_Float16)v;
}

// ---------------------------------------------------------------------------
// C[R x 64] = Asw(R x K, fragment order) * Bsw(K x 64, fragment order), f32 out.
// 4 waves per block, one 16x16 tile per wave; K-loop of 32 per step.
// ---------------------------------------------------------------------------
__global__ void wmma_gemm16(const _Float16* __restrict__ Asw,
                            const _Float16* __restrict__ Bsw,
                            float* __restrict__ C, int R, int K) {
    const int lane = threadIdx.x & 31;
    const int wid  = threadIdx.x >> 5;
    const int rt   = blockIdx.x * 4 + wid;      // row tile
    const int ct   = blockIdx.y;                // col tile (0..3)
    const int KS   = K >> 5;

    const v16h* Ap = (const v16h*)Asw + ((size_t)rt * KS) * 32 + lane;
    const v16h* Bp = (const v16h*)Bsw + (size_t)ct * 32 + lane;

    v8f acc = {};
#pragma unroll 4
    for (int kb = 0; kb < KS; ++kb) {
        v16h a = Ap[(size_t)kb * 32];
        v16h b = Bp[(size_t)kb * 128];
        acc = __builtin_amdgcn_wmma_f32_16x16x32_f16(
            false, a, false, b, (short)0, acc, false, false);
    }

    // C layout (ISA 7.12.2): lane -> col = ct*16+(lane&15); rows = base..base+7
    const int col  = ct * 16 + (lane & 15);
    const int row0 = rt * 16 + ((lane >> 4) << 3);
    float* Cp = C + (size_t)row0 * 64 + col;
#pragma unroll
    for (int v = 0; v < 8; ++v) Cp[(size_t)v * 64] = acc[v];
}

// ------------------------- degree-vector mat-vecs --------------------------
__global__ void matvec_row(const float* __restrict__ A, const float* __restrict__ v,
                           int R, int C, float* __restrict__ out) {
    __shared__ float red[256];
    int r = blockIdx.x;
    float acc = 0.f;
    for (int c = threadIdx.x; c < C; c += 256)
        acc += A[(size_t)r * C + c] * (v ? v[c] : 1.0f);
    red[threadIdx.x] = acc;
    __syncthreads();
    for (int s = 128; s > 0; s >>= 1) {
        if (threadIdx.x < s) red[threadIdx.x] += red[threadIdx.x + s];
        __syncthreads();
    }
    if (threadIdx.x == 0) out[r] = red[0];
}

__global__ void matvec_col(const float* __restrict__ A, const float* __restrict__ v,
                           int R, int C, float* __restrict__ out) {
    int c = blockIdx.x * 256 + threadIdx.x;
    if (c >= C) return;
    float acc = 0.f;
    for (int r = 0; r < R; ++r)
        acc += A[(size_t)r * C + c] * (v ? v[r] : 1.0f);
    out[c] = acc;
}

__global__ void vec_rsqrt(const float* __restrict__ x, const float* __restrict__ y,
                          int n, float* __restrict__ out) {
    int i = blockIdx.x * 256 + threadIdx.x;
    if (i < n) out[i] = rsqrtf(x[i] + (y ? y[i] : 0.f) + EPS);
}

__global__ void vec_inv(const float* __restrict__ x, int n, float* __restrict__ out) {
    int i = blockIdx.x * 256 + threadIdx.x;
    if (i < n) out[i] = 1.0f / (x[i] + EPS);
}

// Mx = dv (.) Z + X   (per-row scale + residual)
__global__ void rowscale_add(const float* __restrict__ Z, const float* __restrict__ dv,
                             const float* __restrict__ X, int R, float* __restrict__ out) {
    int i = blockIdx.x * 256 + threadIdx.x;
    if (i >= R * 64) return;
    int r = i >> 6;
    out[i] = dv[r] * Z[i] + X[i];
}

// Xnext = Mx @ W + b ; also scatter into d_out slice (fp32 FMA: tiny GEMM)
__global__ void proj64(const float* __restrict__ Xin, const float* __restrict__ W,
                       const float* __restrict__ bias, int R,
                       float* __restrict__ Xnext, float* __restrict__ outp,
                       int outLd, int colOff) {
    __shared__ float Ws[64 * 64];
    for (int i = threadIdx.x; i < 64 * 64; i += 256) Ws[i] = W[i];
    __syncthreads();
    int row = blockIdx.x * 4 + (threadIdx.x >> 6);
    int col = threadIdx.x & 63;
    if (row >= R) return;
    float acc = bias[col];
    const float* xr = Xin + (size_t)row * 64;
#pragma unroll 8
    for (int k = 0; k < 64; ++k) acc += xr[k] * Ws[k * 64 + col];
    Xnext[(size_t)row * 64 + col] = acc;
    outp[(size_t)row * outLd + colOff + col] = acc;
}

__global__ void copy_out(const float* __restrict__ src, float* __restrict__ dst,
                         int R, int ld, int colOff) {
    int i = blockIdx.x * 256 + threadIdx.x;
    if (i >= R * 64) return;
    int r = i >> 6, c = i & 63;
    dst[(size_t)r * ld + colOff + c] = src[i];
}

// ---------------------------------------------------------------------------
extern "C" void kernel_launch(void* const* d_in, const int* in_sizes, int n_in,
                              void* d_out, int out_size, void* d_ws, size_t ws_size,
                              hipStream_t stream) {
    const float* H   = (const float*)d_in[0];
    const float* Ue  = (const float*)d_in[1];
    const float* Ie  = (const float*)d_in[2];
    const float* w0  = (const float*)d_in[3];
    const float* b0  = (const float*)d_in[4];
    const float* w1  = (const float*)d_in[5];
    const float* b1  = (const float*)d_in[6];

    float* outU = (float*)d_out;                       // [NU x 192]
    float* outI = outU + (size_t)NU * 192;             // [NI x 192]

    // ---- workspace carve-up (256B aligned) ----
    char* wp = (char*)d_ws;
    auto alloc = [&](size_t bytes) -> void* {
        void* r = (void*)wp;
        wp += (bytes + 255) & ~(size_t)255;
        return r;
    };
    _Float16* Hsw  = (_Float16*)alloc((size_t)NU * NI * 2);   // H  fragment order
    _Float16* Htsw = (_Float16*)alloc((size_t)NU * NI * 2);   // H^T fragment order
    _Float16* Bsw  = (_Float16*)alloc((size_t)NU * 64 * 2);
    size_t fb = (size_t)NU * 64 * 4;                          // max skinny buffer
    float* Y1  = (float*)alloc(fb);
    float* Y2  = (float*)alloc(fb);
    float* Tt  = (float*)alloc(fb);
    float* U1  = (float*)alloc(fb);
    float* U2  = (float*)alloc(fb);
    float* Zb  = (float*)alloc(fb);
    float* Mx  = (float*)alloc(fb);
    float* Xua = (float*)alloc(fb);
    float* Xub = (float*)alloc(fb);
    float* Xia = (float*)alloc(fb);
    float* Xib = (float*)alloc(fb);
    float* rs  = (float*)alloc(NU * 4);   // H 1
    float* cs  = (float*)alloc(NU * 4);   // H^T 1
    float* av  = (float*)alloc(NU * 4);   // H^T r
    float* bv  = (float*)alloc(NU * 4);   // H a
    float* pv  = (float*)alloc(NU * 4);   // H c
    float* qv  = (float*)alloc(NU * 4);   // H^T p
    float* dvu   = (float*)alloc(NU * 4);
    float* dvi   = (float*)alloc(NU * 4);
    float* de2au = (float*)alloc(NU * 4);
    float* de2bu = (float*)alloc(NU * 4);
    float* de2ai = (float*)alloc(NU * 4);
    float* de2bi = (float*)alloc(NU * 4);

    const int T = 256;
    auto blocks = [](size_t n) { return (unsigned)((n + 255) / 256); };

    // ---- one-time: swizzle H and H^T into WMMA A-fragment layout ----
    size_t nH = (size_t)NU * NI;
    make_a_frag<<<blocks(nH), T, 0, stream>>>(H, NU, NI, NI, 0, Hsw);
    make_a_frag<<<blocks(nH), T, 0, stream>>>(H, NI, NU, NI, 1, Htsw);

    // ---- one-time: degree vectors (Hu row/col sums factor through H) ----
    matvec_row<<<NU, T, 0, stream>>>(H, nullptr, NU, NI, rs);   // r = H 1
    matvec_col<<<blocks(NI), T, 0, stream>>>(H, nullptr, NU, NI, cs); // c = H^T 1
    matvec_col<<<blocks(NI), T, 0, stream>>>(H, rs, NU, NI, av);      // a = H^T r
    matvec_row<<<NU, T, 0, stream>>>(H, av, NU, NI, bv);              // b = H a
    matvec_row<<<NU, T, 0, stream>>>(H, cs, NU, NI, pv);              // p = H c
    matvec_col<<<blocks(NI), T, 0, stream>>>(H, pv, NU, NI, qv);      // q = H^T p

    vec_rsqrt<<<blocks(NU), T, 0, stream>>>(rs, bv, NU, dvu);  // dv_u = (r+b)^-1/2
    vec_rsqrt<<<blocks(NI), T, 0, stream>>>(cs, qv, NI, dvi);  // dv_i = (c+q)^-1/2
    vec_inv  <<<blocks(NI), T, 0, stream>>>(cs, NI, de2au);    // de_u^2 part1
    vec_inv  <<<blocks(NI), T, 0, stream>>>(qv, NI, de2bu);    // de_u^2 part2
    vec_inv  <<<blocks(NU), T, 0, stream>>>(rs, NU, de2ai);    // de_i^2 part1
    vec_inv  <<<blocks(NU), T, 0, stream>>>(bv, NU, de2bi);    // de_i^2 part2

    // ---- layer-0 embeddings straight to output ----
    copy_out<<<blocks((size_t)NU * 64), T, 0, stream>>>(Ue, outU, NU, 192, 0);
    copy_out<<<blocks((size_t)NI * 64), T, 0, stream>>>(Ie, outI, NI, 192, 0);

    auto gemm = [&](const _Float16* A, int R, int K, float* C) {
        dim3 g(R / 64, 4);
        wmma_gemm16<<<g, 128, 0, stream>>>(A, Bsw, C, R, K);
    };
    auto bfrag = [&](const float* X, const float* Y, const float* s, int K) {
        make_b_frag<<<blocks((size_t)K * 64), T, 0, stream>>>(X, Y, s, K, Bsw);
    };

    // One hypergraph-propagation channel:
    //   Y1 = At (dv.X) ; Y2 = At (Am Y1)
    //   Z  = Am (de2a.Y1 + At (Am (de2b.Y2)))
    //   Xnext = (dv.Z + X) @ W + bias
    auto channel = [&](const _Float16* Am, const _Float16* At, int n, int m,
                       const float* Xcur, const float* dv, const float* de2a,
                       const float* de2b, const float* W, const float* Bias,
                       float* Xnext, float* outBase, int colOff) {
        bfrag(Xcur, nullptr, dv, n);      gemm(At, m, n, Y1);
        bfrag(Y1, nullptr, nullptr, m);   gemm(Am, n, m, Tt);
        bfrag(Tt, nullptr, nullptr, n);   gemm(At, m, n, Y2);
        bfrag(Y2, nullptr, de2b, m);      gemm(Am, n, m, U1);
        bfrag(U1, nullptr, nullptr, n);   gemm(At, m, n, U2);
        bfrag(Y1, U2, de2a, m);           gemm(Am, n, m, Zb);
        rowscale_add<<<blocks((size_t)n * 64), T, 0, stream>>>(Zb, dv, Xcur, n, Mx);
        proj64<<<n / 4, T, 0, stream>>>(Mx, W, Bias, n, Xnext, outBase, 192, colOff);
    };

    // layer 0 (both channels read the pre-layer embeddings)
    channel(Hsw, Htsw, NU, NI, Ue,  dvu, de2au, de2bu, w0, b0, Xua, outU, 64);
    channel(Htsw, Hsw, NI, NU, Ie,  dvi, de2ai, de2bi, w0, b0, Xia, outI, 64);
    // layer 1
    channel(Hsw, Htsw, NU, NI, Xua, dvu, de2au, de2bu, w1, b1, Xub, outU, 128);
    channel(Htsw, Hsw, NI, NU, Xia, dvi, de2ai, de2bi, w1, b1, Xib, outI, 128);

    (void)in_sizes; (void)n_in; (void)out_size; (void)ws_size;
}
